// MPNNPlusPredictor_48095043781133
// MI455X (gfx1250) — matrix operand order repaired and checked by reading, not compiled, not run on hardware
//
#include <hip/hip_runtime.h>
#include <hip/hip_bf16.h>

#define DIM 32
#define HID 128
#define NN  8192
#define EE  131072
#define BB  64

typedef __attribute__((ext_vector_type(16))) _Float16 v16h;
typedef __attribute__((ext_vector_type(2)))  _Float16 h2v;
typedef __attribute__((ext_vector_type(8)))  float    v8f;

union V16 {
    v16h v;
    h2v  p[8];
    uint4 q[2];
    _Float16 h[16];
};

__device__ __forceinline__ float sigf(float x) { return 1.0f / (1.0f + __expf(-x)); }

// ---------------------------------------------------------------- utility
__global__ void k_zero(float* __restrict__ p, int n) {
    int i = blockIdx.x * blockDim.x + threadIdx.x;
    if (i < n) p[i] = 0.0f;
}

// deg[dst] += 1
__global__ void k_deg(const int* __restrict__ eidx, float* __restrict__ deg) {
    int e = blockIdx.x * blockDim.x + threadIdx.x;
    if (e < EE) atomicAdd(&deg[eidx[EE + e]], 1.0f);
}

// out = relu(x @ lin0_w + lin0_b), fp32 + fp16 copies
__global__ void k_lin0(const float* __restrict__ x, const float* __restrict__ w,
                       const float* __restrict__ b, float* __restrict__ outf,
                       _Float16* __restrict__ outh) {
    int i = blockIdx.x * blockDim.x + threadIdx.x;
    if (i >= NN * DIM) return;
    int n = i >> 5, d = i & 31;
    float v = x[n] * w[d] + b[d];
    v = v > 0.0f ? v : 0.0f;
    outf[i] = v;
    outh[i] = (_Float16)v;
}

// h = relu(edge_attr @ enn_w1 + enn_b1) as fp16 [E,128]
__global__ void k_edgeh(const float* __restrict__ ea, const float* __restrict__ w1,
                        const float* __restrict__ b1, _Float16* __restrict__ h) {
    int i = blockIdx.x * blockDim.x + threadIdx.x;
    if (i >= EE * HID) return;
    int e = i >> 7, k = i & 127;
    float v = ea[e * 2] * w1[k] + ea[e * 2 + 1] * w1[HID + k] + b1[k];
    v = v > 0.0f ? v : 0.0f;
    h[i] = (_Float16)v;
}

// transpose enn_w2 [128][d*32+o] -> w2t [k][o][d] fp16 ; b2t [o][d] fp16
__global__ void k_w2t(const float* __restrict__ w2, const float* __restrict__ b2,
                      _Float16* __restrict__ w2t, _Float16* __restrict__ b2t) {
    int i = blockIdx.x * blockDim.x + threadIdx.x;
    if (i < HID * 1024) {
        int k = i >> 10;
        int rem = i & 1023;
        int d = rem >> 5, o = rem & 31;
        w2t[(k * DIM + o) * DIM + d] = (_Float16)w2[i];
    } else if (i < HID * 1024 + 1024) {
        int j = i - HID * 1024;
        int d = j >> 5, o = j & 31;
        b2t[o * DIM + d] = (_Float16)b2[d * DIM + o];
    }
}

// ---------------------------------------------------------------- WMMA message kernel
// msg[16,32] per tile = P[16,4096] @ W2r[4096,32] + Out[16,32] @ B2r
// with P[e, k*32+d] = h[e,k]*out[e,d].  Scatter-add into agg via f32 atomics.
#define SMEM_W2   (HID * 1024 * 2)          // 262144 B
#define SMEM_B2   (1024 * 2)                // 2048 B
#define HROW      136                       // padded half-stride (bank-safe)
#define SMEM_HW   (16 * HROW * 2)           // per-wave h tile bytes (4352)
#define SMEM_MSG  (SMEM_W2 + SMEM_B2 + 8 * SMEM_HW)

__global__ void __launch_bounds__(256)
k_msg(const _Float16* __restrict__ hbuf,   // [E,128] fp16
      const _Float16* __restrict__ w2t,    // [128*32*32] (k,o,d)
      const _Float16* __restrict__ b2t,    // [32*32] (o,d)
      const _Float16* __restrict__ outh,   // [N,32] fp16
      const int* __restrict__ eidx,        // [2,E]
      float* __restrict__ agg)             // [N,32]
{
    extern __shared__ char smem[];
    _Float16* w2s = (_Float16*)smem;
    _Float16* b2s = (_Float16*)(smem + SMEM_W2);
    _Float16* hs  = (_Float16*)(smem + SMEM_W2 + SMEM_B2);

    const int tid = threadIdx.x;
    const int lane = tid & 31;
    const int wave = tid >> 5;
    const int halfsel = lane >> 4;     // 0 (lanes 0-15) / 1 (lanes 16-31)
    const int nIdx = lane & 15;

    // cooperative fill of W2 (fp16, transposed) + bias into LDS
    {
        const uint4* gs = (const uint4*)w2t;
        uint4* gd = (uint4*)w2s;
        for (int i = tid; i < SMEM_W2 / 16; i += 256) gd[i] = gs[i];
        const uint4* bs = (const uint4*)b2t;
        uint4* bd = (uint4*)b2s;
        for (int i = tid; i < SMEM_B2 / 16; i += 256) bd[i] = bs[i];
    }
    __syncthreads();

    _Float16* hrow = hs + wave * (16 * HROW);
    const int waveGlobal = blockIdx.x * 8 + wave;
    const int stride = gridDim.x * 8;

    for (int t = waveGlobal; t < (EE / 16); t += stride) {
        const int e0 = t * 16;
        const int m = nIdx;            // edge row handled by this lane
        const int e = e0 + m;

        // stage h rows (16 edges x 128 halfs) into padded LDS rows
        {
            const uint4* src = (const uint4*)(hbuf + (size_t)e * HID + halfsel * 64);
            uint4* dst = (uint4*)(hrow + m * HROW + halfsel * 64);
#pragma unroll
            for (int j = 0; j < 8; ++j) dst[j] = src[j];
        }

        // A-layout gather of source-node features (fp16), contiguous k = 16*halfsel + s, same order as B
        const int s = eidx[e];
        const int K0a = halfsel * 16;
        V16 of;
        of.q[0] = *(const uint4*)(outh + (size_t)s * DIM + K0a);
        of.q[1] = *(const uint4*)(outh + (size_t)s * DIM + K0a + 8);

        __syncthreads();

        v8f acc0 = {};
        v8f acc1 = {};
        const int K0b = halfsel * 16;

        for (int k = 0; k < HID; ++k) {
            _Float16 hv = hrow[m * HROW + k];
            h2v hsplat;
            hsplat[0] = hv; hsplat[1] = hv;
            V16 a;
#pragma unroll
            for (int j = 0; j < 8; ++j) a.p[j] = of.p[j] * hsplat;

            const _Float16* b0p = w2s + ((k * DIM + nIdx) * DIM + K0b);
            const _Float16* b1p = w2s + ((k * DIM + 16 + nIdx) * DIM + K0b);
            V16 b0, b1;
            b0.q[0] = ((const uint4*)b0p)[0];
            b0.q[1] = ((const uint4*)b0p)[1];
            b1.q[0] = ((const uint4*)b1p)[0];
            b1.q[1] = ((const uint4*)b1p)[1];

            acc0 = __builtin_amdgcn_wmma_f32_16x16x32_f16(false, a.v, false, b0.v,
                                                          (short)0, acc0, false, false);
            acc1 = __builtin_amdgcn_wmma_f32_16x16x32_f16(false, a.v, false, b1.v,
                                                          (short)0, acc1, false, false);
        }

        // bias chunk: A = out fragment, B = reshaped enn_b2
        {
            const _Float16* b0p = b2s + (nIdx * DIM + K0b);
            const _Float16* b1p = b2s + ((16 + nIdx) * DIM + K0b);
            V16 b0, b1;
            b0.q[0] = ((const uint4*)b0p)[0];
            b0.q[1] = ((const uint4*)b0p)[1];
            b1.q[0] = ((const uint4*)b1p)[0];
            b1.q[1] = ((const uint4*)b1p)[1];
            acc0 = __builtin_amdgcn_wmma_f32_16x16x32_f16(false, of.v, false, b0.v,
                                                          (short)0, acc0, false, false);
            acc1 = __builtin_amdgcn_wmma_f32_16x16x32_f16(false, of.v, false, b1.v,
                                                          (short)0, acc1, false, false);
        }

        // scatter: D tile row M = edge index in tile, col N = output dim
        const int mbase = (lane < 16) ? 0 : 8;
#pragma unroll
        for (int r = 0; r < 8; ++r) {
            const int me = mbase + r;
            const int node = eidx[EE + e0 + me];
            atomicAdd(&agg[node * DIM + nIdx], acc0[r]);
            atomicAdd(&agg[node * DIM + 16 + nIdx], acc1[r]);
        }
        __syncthreads();
    }
}

// ---------------------------------------------------------------- node update (NNConv tail + GRU)
__global__ void __launch_bounds__(128)
k_node(const float* __restrict__ agg, const float* __restrict__ deg,
       const float* __restrict__ rootw, const float* __restrict__ convb,
       const float* __restrict__ wih, const float* __restrict__ whh,
       const float* __restrict__ bih, const float* __restrict__ bhh,
       float* __restrict__ outf, _Float16* __restrict__ outh)
{
    int n = blockIdx.x * blockDim.x + threadIdx.x;
    if (n >= NN) return;
    float xr[DIM], mr[DIM];
    float dg = deg[n];
    dg = dg < 1.0f ? 1.0f : dg;
#pragma unroll
    for (int d = 0; d < DIM; ++d) xr[d] = outf[n * DIM + d];
    for (int o = 0; o < DIM; ++o) {
        float s = convb[o] + agg[n * DIM + o] / dg;
#pragma unroll
        for (int d = 0; d < DIM; ++d) s += xr[d] * rootw[d * DIM + o];
        mr[o] = s > 0.0f ? s : 0.0f;
    }
    for (int d = 0; d < DIM; ++d) {
        float gir = bih[d], giz = bih[DIM + d], gin = bih[2 * DIM + d];
        float ghr = bhh[d], ghz = bhh[DIM + d], ghn = bhh[2 * DIM + d];
#pragma unroll
        for (int u = 0; u < DIM; ++u) {
            float mu = mr[u], hu = xr[u];
            gir += mu * wih[u * 96 + d];            ghr += hu * whh[u * 96 + d];
            giz += mu * wih[u * 96 + DIM + d];      ghz += hu * whh[u * 96 + DIM + d];
            gin += mu * wih[u * 96 + 2 * DIM + d];  ghn += hu * whh[u * 96 + 2 * DIM + d];
        }
        float r = sigf(gir + ghr);
        float z = sigf(giz + ghz);
        float nn_ = tanhf(gin + r * ghn);
        float hnew = (1.0f - z) * nn_ + z * xr[d];
        outf[n * DIM + d] = hnew;
        outh[n * DIM + d] = (_Float16)hnew;
    }
}

// ---------------------------------------------------------------- set2set helpers
__global__ void k_bounds(const int* __restrict__ batch, int* __restrict__ starts,
                         int* __restrict__ ends) {
    int i = blockIdx.x * blockDim.x + threadIdx.x;
    if (i >= NN) return;
    int b = batch[i];
    if (i == 0 || batch[i - 1] != b) starts[b] = i;
    if (i == NN - 1 || batch[i + 1] != b) ends[b] = i + 1;
}

__global__ void __launch_bounds__(256)
k_lstm(const float* __restrict__ wih, const float* __restrict__ whh,
       const float* __restrict__ bihp, const float* __restrict__ bhhp,
       float* __restrict__ qh, float* __restrict__ qc, float* __restrict__ qstar)
{
    __shared__ float gates[BB * 4 * DIM];   // 64 x 128
    int tid = threadIdx.x;
    for (int idx = tid; idx < BB * 128; idx += 256) {
        int g = idx >> 7, j = idx & 127;
        float s = bihp[j] + bhhp[j];
        const float* qs = qstar + g * 2 * DIM;
#pragma unroll
        for (int u = 0; u < 2 * DIM; ++u) s += qs[u] * wih[u * 128 + j];
        const float* qhg = qh + g * DIM;
#pragma unroll
        for (int u = 0; u < DIM; ++u) s += qhg[u] * whh[u * 128 + j];
        gates[idx] = s;
    }
    __syncthreads();
    for (int idx = tid; idx < BB * DIM; idx += 256) {
        int g = idx >> 5, d = idx & 31;
        const float* gg = gates + g * 128;
        float i_ = sigf(gg[d]);
        float f_ = sigf(gg[DIM + d]);
        float g_ = tanhf(gg[2 * DIM + d]);
        float o_ = sigf(gg[3 * DIM + d]);
        float c = f_ * qc[idx] + i_ * g_;
        qc[idx] = c;
        float hh = o_ * tanhf(c);
        qh[idx] = hh;
        qstar[g * 2 * DIM + d] = hh;
    }
}

__global__ void __launch_bounds__(256)
k_attn(const float* __restrict__ outf, const int* __restrict__ starts,
       const int* __restrict__ ends, const float* __restrict__ qh,
       float* __restrict__ ews, float* __restrict__ qstar)
{
    int g = blockIdx.x;
    int tid = threadIdx.x;
    __shared__ float qs[DIM];
    __shared__ float red[256];
    __shared__ float rvec[DIM];
    if (tid < DIM) {
        qs[tid] = qh[g * DIM + tid];
        rvec[tid] = 0.0f;
    }
    __syncthreads();
    int s0 = starts[g], s1 = ends[g];

    float lmax = -3.4e38f;
    for (int i = s0 + tid; i < s1; i += 256) {
        float e = 0.0f;
#pragma unroll
        for (int d = 0; d < DIM; ++d) e += outf[i * DIM + d] * qs[d];
        ews[i] = e;
        lmax = fmaxf(lmax, e);
    }
    red[tid] = lmax;
    __syncthreads();
    for (int st = 128; st > 0; st >>= 1) {
        if (tid < st) red[tid] = fmaxf(red[tid], red[tid + st]);
        __syncthreads();
    }
    float gmax = red[0];
    __syncthreads();

    float lsum = 0.0f;
    float lr[DIM];
#pragma unroll
    for (int d = 0; d < DIM; ++d) lr[d] = 0.0f;
    for (int i = s0 + tid; i < s1; i += 256) {
        float a = __expf(ews[i] - gmax);
        lsum += a;
#pragma unroll
        for (int d = 0; d < DIM; ++d) lr[d] += a * outf[i * DIM + d];
    }
    red[tid] = lsum;
    __syncthreads();
    for (int st = 128; st > 0; st >>= 1) {
        if (tid < st) red[tid] += red[tid + st];
        __syncthreads();
    }
    float denom = red[0];
#pragma unroll
    for (int d = 0; d < DIM; ++d) atomicAdd(&rvec[d], lr[d]);
    __syncthreads();
    if (tid < DIM) {
        float r = (denom > 0.0f) ? rvec[tid] / denom : 0.0f;
        qstar[g * 2 * DIM + DIM + tid] = r;
    }
}

__global__ void k_final(const float* __restrict__ qstar, const float* __restrict__ w1,
                        const float* __restrict__ b1, const float* __restrict__ w2,
                        const float* __restrict__ b2, float* __restrict__ y)
{
    int g = blockIdx.x * blockDim.x + threadIdx.x;
    if (g >= BB) return;
    float acc = b2[0];
    for (int o = 0; o < DIM; ++o) {
        float s = b1[o];
#pragma unroll
        for (int u = 0; u < 2 * DIM; ++u) s += qstar[g * 2 * DIM + u] * w1[u * DIM + o];
        s = s > 0.0f ? s : 0.0f;
        acc += s * w2[o];
    }
    y[g] = acc;
}

// ---------------------------------------------------------------- launch
extern "C" void kernel_launch(void* const* d_in, const int* in_sizes, int n_in,
                              void* d_out, int out_size, void* d_ws, size_t ws_size,
                              hipStream_t stream) {
    (void)in_sizes; (void)n_in; (void)out_size; (void)ws_size;

    const float* x         = (const float*)d_in[0];
    const float* edge_attr = (const float*)d_in[1];
    const float* lin0_w    = (const float*)d_in[2];
    const float* lin0_b    = (const float*)d_in[3];
    const float* enn_w1    = (const float*)d_in[4];
    const float* enn_b1    = (const float*)d_in[5];
    const float* enn_w2    = (const float*)d_in[6];
    const float* enn_b2    = (const float*)d_in[7];
    const float* root_w    = (const float*)d_in[8];
    const float* conv_b    = (const float*)d_in[9];
    const float* gru_wih   = (const float*)d_in[10];
    const float* gru_whh   = (const float*)d_in[11];
    const float* gru_bih   = (const float*)d_in[12];
    const float* gru_bhh   = (const float*)d_in[13];
    const float* s2s_wih   = (const float*)d_in[14];
    const float* s2s_whh   = (const float*)d_in[15];
    const float* s2s_bih   = (const float*)d_in[16];
    const float* s2s_bhh   = (const float*)d_in[17];
    const float* lin1_w    = (const float*)d_in[18];
    const float* lin1_b    = (const float*)d_in[19];
    const float* lin2_w    = (const float*)d_in[20];
    const float* lin2_b    = (const float*)d_in[21];
    const int* edge_index  = (const int*)d_in[22];
    const int* batch       = (const int*)d_in[23];
    float* out = (float*)d_out;

    // workspace carving (deterministic)
    char* p = (char*)d_ws;
    auto carve = [&](size_t bytes) -> char* {
        char* r = p;
        p += (bytes + 255) & ~(size_t)255;
        return r;
    };
    _Float16* hbuf = (_Float16*)carve((size_t)EE * HID * 2);   // 32 MB
    _Float16* w2t  = (_Float16*)carve((size_t)HID * 1024 * 2); // 256 KB
    _Float16* b2t  = (_Float16*)carve(1024 * 2);
    float* outf    = (float*)carve((size_t)NN * DIM * 4);
    _Float16* outh = (_Float16*)carve((size_t)NN * DIM * 2);
    float* agg     = (float*)carve((size_t)NN * DIM * 4);
    float* deg     = (float*)carve((size_t)NN * 4);
    float* ews     = (float*)carve((size_t)NN * 4);
    float* s2sbuf  = (float*)carve((size_t)BB * DIM * 4 * 4);  // qh | qc | qstar
    float* qh    = s2sbuf;
    float* qc    = s2sbuf + BB * DIM;
    float* qstar = s2sbuf + 2 * BB * DIM;
    int* starts = (int*)carve(BB * 4);
    int* ends   = (int*)carve(BB * 4);

    // prep
    k_zero<<<(NN + 255) / 256, 256, 0, stream>>>(deg, NN);
    k_zero<<<1, 64, 0, stream>>>((float*)starts, BB);
    k_zero<<<1, 64, 0, stream>>>((float*)ends, BB);
    k_deg<<<EE / 256, 256, 0, stream>>>(edge_index, deg);
    k_lin0<<<(NN * DIM) / 256, 256, 0, stream>>>(x, lin0_w, lin0_b, outf, outh);
    k_edgeh<<<(EE * HID) / 256, 256, 0, stream>>>(edge_attr, enn_w1, enn_b1, hbuf);
    k_w2t<<<(HID * 1024 + 1024) / 256, 256, 0, stream>>>(enn_w2, enn_b2, w2t, b2t);

    // message-passing layers (fused WMMA message + scatter, then GRU update)
    for (int l = 0; l < 3; ++l) {
        k_zero<<<(NN * DIM) / 256, 256, 0, stream>>>(agg, NN * DIM);
        k_msg<<<256, 256, SMEM_MSG, stream>>>(hbuf, w2t, b2t, outh, edge_index, agg);
        k_node<<<NN / 128, 128, 0, stream>>>(agg, deg, root_w, conv_b, gru_wih, gru_whh,
                                             gru_bih, gru_bhh, outf, outh);
    }

    // set2set pooling
    k_zero<<<(BB * DIM * 4 + 255) / 256, 256, 0, stream>>>(s2sbuf, BB * DIM * 4);
    k_bounds<<<NN / 256, 256, 0, stream>>>(batch, starts, ends);
    for (int s = 0; s < 3; ++s) {
        k_lstm<<<1, 256, 0, stream>>>(s2s_wih, s2s_whh, s2s_bih, s2s_bhh, qh, qc, qstar);
        k_attn<<<BB, 256, 0, stream>>>(outf, starts, ends, qh, ews, qstar);
    }

    // readout MLP
    k_final<<<1, 64, 0, stream>>>(qstar, lin1_w, lin1_b, lin2_w, lin2_b, out);
}
